// LaneAtt_28080496181785
// MI455X (gfx1250) — matrix operand order repaired and checked by reading, not compiled, not run on hardware
//
#include <hip/hip_runtime.h>
#include <hip/hip_bf16.h>

typedef _Float16 h8   __attribute__((ext_vector_type(8)));
typedef _Float16 v16h __attribute__((ext_vector_type(16)));
typedef float    v8f  __attribute__((ext_vector_type(8)));

#define FEAT_H 11
#define FEAT_W 20
#define C_BB   512
#define C_FEAT 64
#define N_ANCH 2784          // 432 + 1920 + 432 anchors, = 174*16 = 87*32 = 29*96
#define D_FEAT 704           // C_FEAT * FEAT_H, = 22*32
#define D_CAT  1408          // 2*D_FEAT, = 44*32
#define NCR    96            // cls(2)+reg(73) padded to 3*32
#define BATCH  8

// ---------------------------------------------------------------- conv1 (1x1)
__global__ void conv1_kernel(const float* __restrict__ x, const float* __restrict__ w,
                             const float* __restrict__ bias, float* __restrict__ feat) {
    __shared__ float xs[C_BB];
    int pix = blockIdx.x;                       // b*220 + h*20 + wcol
    int b   = pix / (FEAT_H * FEAT_W);
    int hw  = pix % (FEAT_H * FEAT_W);
    const float* xp = x + (size_t)b * C_BB * FEAT_H * FEAT_W + hw;
    for (int c = threadIdx.x; c < C_BB; c += blockDim.x)
        xs[c] = xp[(size_t)c * FEAT_H * FEAT_W];
    __syncthreads();
    int o = threadIdx.x;                        // blockDim == C_FEAT == 64
    const float* wr = w + o * C_BB;
    float acc = bias[o];
#pragma unroll 8
    for (int c = 0; c < C_BB; ++c) acc = fmaf(xs[c], wr[c], acc);
    feat[(size_t)(b * C_FEAT + o) * FEAT_H * FEAT_W + hw] = acc;
}

// ------------------------------------------------- ROI gather -> baf(f16) x2
// cat_h layout: [B, N, D_CAT]; cols [704..1408) = baf. bafT: [B, D_FEAT, N].
__global__ void gather_kernel(const float* __restrict__ feat,
                              const int* __restrict__ cut_xs,
                              const unsigned char* __restrict__ invalid,
                              _Float16* __restrict__ cat_h,
                              _Float16* __restrict__ bafT) {
    int bn = blockIdx.x;
    int b = bn / N_ANCH, n = bn % N_ANCH;
    for (int d = threadIdx.x; d < D_FEAT; d += blockDim.x) {
        int c = d / FEAT_H, h = d % FEAT_H;     // d = c*Hf + h (reshape order)
        int xw = cut_xs[n * FEAT_H + h];
        float v = invalid[n * FEAT_H + h] ? 0.f
                : feat[(((size_t)b * C_FEAT + c) * FEAT_H + h) * FEAT_W + xw];
        cat_h[(size_t)bn * D_CAT + D_FEAT + d]       = (_Float16)v;
        bafT[((size_t)b * D_FEAT + d) * N_ANCH + n]  = (_Float16)v;
    }
}

// --------------------------------------------------------- weight prep (f16)
__global__ void prep_attn_kernel(const float* __restrict__ attn_w,
                                 const float* __restrict__ attn_b,
                                 _Float16* __restrict__ attn_wh,
                                 float* __restrict__ attn_bp) {
    int i = blockIdx.x * blockDim.x + threadIdx.x;
    if (i < N_ANCH * D_FEAT) {
        int r = i / D_FEAT;
        attn_wh[i] = (r < N_ANCH - 1) ? (_Float16)attn_w[i] : (_Float16)0.f;
    }
    if (i < N_ANCH) attn_bp[i] = (i < N_ANCH - 1) ? attn_b[i] : 0.f;
}

__global__ void prep_cr_kernel(const float* __restrict__ cls_w, const float* __restrict__ cls_b,
                               const float* __restrict__ reg_w, const float* __restrict__ reg_b,
                               _Float16* __restrict__ wcr, float* __restrict__ bcr) {
    int i = blockIdx.x * blockDim.x + threadIdx.x;
    if (i < NCR * D_CAT) {
        int r = i / D_CAT, k = i % D_CAT;
        float v = 0.f;
        if (r < 2)       v = cls_w[r * D_CAT + k];
        else if (r < 75) v = reg_w[(r - 2) * D_CAT + k];
        wcr[i] = (_Float16)v;
    }
    if (i < NCR) {
        float v = 0.f;
        if (i < 2) v = cls_b[i]; else if (i < 75) v = reg_b[i - 2];
        bcr[i] = v;
    }
}

// ------------------------------------------------------ generic NT WMMA GEMM
// C[m,n] = sum_k A[m*LDA+k] * B[n*LDB+k] (+ bias[n]).
// One wave computes a (MT*16) x 32 tile: MT A-frags reuse 2 B-frags per K-step
// -> 12.0 MAC/byte of fragment traffic at MT=6 (vs 5.3 for a 16x32 tile).
// Strides are compile-time so every mi/kb fragment address is a 24-bit
// instruction immediate off ONE A base pointer and ONE B base pointer.
// __launch_bounds__(256, 1): relax the 8-wave/SIMD occupancy floor so the
// allocator can take ~144 VGPRs (96 accum + 16 B + 8 A + addresses) without
// spilling an accumulator; 2-4 resident waves/SIMD are enough for this
// WMMA-throughput-bound loop.
// A-frag: lane holds row (lane&15), K = lhalf*16 + {0..15}
// B-frag: lane holds col (lane&15), K = lhalf*16 + {0..15}  (contiguous 32B)
template <bool F16OUT, int MT, int LDA, int LDB, int LDC>
__global__ void __launch_bounds__(256, 1)
gemm_nt_wmma(const _Float16* __restrict__ A,
             const _Float16* __restrict__ B,
             void* __restrict__ Cout,
             int MtilesMT, int Ntiles2, int K,
             const float* __restrict__ bias) {
    int wid = blockIdx.x * (blockDim.x >> 5) + (threadIdx.x >> 5);
    if (wid >= MtilesMT * Ntiles2) return;      // uniform per-wave exit
    int tm = wid / Ntiles2, tn = wid % Ntiles2;
    int lane  = threadIdx.x & 31;
    int lhalf = lane >> 4;
    int lmod  = lane & 15;

    v8f acc[MT][2];
#pragma unroll
    for (int mi = 0; mi < MT; ++mi) { acc[mi][0] = {}; acc[mi][1] = {}; }

    const _Float16* aptr = A + (size_t)(tm * MT * 16 + lmod) * LDA + lhalf * 16;
    const _Float16* bptr = B + (size_t)(tn * 32 + lmod) * LDB + lhalf * 16;

    for (int kb = 0; kb < K; kb += 32) {
        v16h b0 = *(const v16h*)(bptr);
        v16h b1 = *(const v16h*)(bptr + 16 * LDB);      // immediate offset
#pragma unroll
        for (int mi = 0; mi < MT; ++mi) {
            union { v16h v; h8 h[2]; } a;
            a.h[0] = *(const h8*)(aptr + mi * 16 * LDA);        // immediate
            a.h[1] = *(const h8*)(aptr + mi * 16 * LDA + 8);    // immediate
            acc[mi][0] = __builtin_amdgcn_wmma_f32_16x16x32_f16(false, a.v, false, b0,
                                                                (short)0, acc[mi][0], false, false);
            acc[mi][1] = __builtin_amdgcn_wmma_f32_16x16x32_f16(false, a.v, false, b1,
                                                                (short)0, acc[mi][1], false, false);
        }
        aptr += 32;
        bptr += 32;
    }

    int col0 = tn * 32 + lmod;
    float bb0 = bias ? bias[col0]      : 0.f;
    float bb1 = bias ? bias[col0 + 16] : 0.f;
    int rowbase = tm * MT * 16 + lhalf * 8;
#pragma unroll
    for (int mi = 0; mi < MT; ++mi) {
#pragma unroll
        for (int r = 0; r < 8; ++r) {
            size_t idx = (size_t)rowbase * LDC + (size_t)(mi * 16 + r) * LDC + col0;
            if (F16OUT) {
                ((_Float16*)Cout)[idx]      = (_Float16)(acc[mi][0][r] + bb0);
                ((_Float16*)Cout)[idx + 16] = (_Float16)(acc[mi][1][r] + bb1);
            } else {
                ((float*)Cout)[idx]      = acc[mi][0][r] + bb0;
                ((float*)Cout)[idx + 16] = acc[mi][1][r] + bb1;
            }
        }
    }
}

// -------------------------------- softmax over 2783 cols + diagonal scatter
// P[n][j] = 0 if j==n else softmax(S[n])[j - (j>n)]; written as f16.
__global__ void softmax_shift_kernel(const float* __restrict__ S, _Float16* __restrict__ P) {
    int n = blockIdx.x;
    const float* srow = S + (size_t)n * N_ANCH;
    __shared__ float red[256];
    int tid = threadIdx.x;

    float mx = -1e30f;
    for (int k = tid; k < N_ANCH - 1; k += 256) mx = fmaxf(mx, srow[k]);
    red[tid] = mx; __syncthreads();
    for (int s = 128; s > 0; s >>= 1) {
        if (tid < s) red[tid] = fmaxf(red[tid], red[tid + s]);
        __syncthreads();
    }
    mx = red[0]; __syncthreads();

    float sum = 0.f;
    for (int k = tid; k < N_ANCH - 1; k += 256) sum += __expf(srow[k] - mx);
    red[tid] = sum; __syncthreads();
    for (int s = 128; s > 0; s >>= 1) {
        if (tid < s) red[tid] += red[tid + s];
        __syncthreads();
    }
    float inv = 1.f / red[0];

    _Float16* prow = P + (size_t)n * N_ANCH;
    for (int j = tid; j < N_ANCH; j += 256) {
        float v = 0.f;
        if (j != n) {
            int k = j - (j > n ? 1 : 0);
            v = __expf(srow[k] - mx) * inv;
        }
        prow[j] = (_Float16)v;
    }
}

// ----------------------------------------------------------- final assembly
__global__ void assemble_kernel(const float* __restrict__ cr,
                                const float* __restrict__ anchors,
                                float* __restrict__ out) {
    int bn = blockIdx.x;
    int n  = bn % N_ANCH;
    int j  = threadIdx.x;
    if (j >= 77) return;
    const float* crr = cr + (size_t)bn * NCR;
    float v;
    if (j < 2)      v = crr[j];
    else if (j < 4) v = anchors[n * 77 + j];
    else            v = anchors[n * 77 + j] + crr[2 + (j - 4)];
    out[(size_t)bn * 77 + j] = v;
}

extern "C" void kernel_launch(void* const* d_in, const int* in_sizes, int n_in,
                              void* d_out, int out_size, void* d_ws, size_t ws_size,
                              hipStream_t stream) {
    const float* x        = (const float*)d_in[0];
    const float* conv1_w  = (const float*)d_in[1];
    const float* conv1_b  = (const float*)d_in[2];
    const float* attn_w   = (const float*)d_in[3];
    const float* attn_b   = (const float*)d_in[4];
    const float* cls_w    = (const float*)d_in[5];
    const float* cls_b    = (const float*)d_in[6];
    const float* reg_w    = (const float*)d_in[7];
    const float* reg_b    = (const float*)d_in[8];
    const float* anchors  = (const float*)d_in[9];
    const int*   cut_xs   = (const int*)d_in[10];
    const unsigned char* invalid = (const unsigned char*)d_in[11];
    float* out = (float*)d_out;

    char* ws = (char*)d_ws;
    size_t off = 0;
    auto carve = [&](size_t bytes) -> char* {
        char* p = ws + off;
        off += (bytes + 255) & ~(size_t)255;
        return p;
    };

    float*    feat     = (float*)   carve(sizeof(float) * BATCH * C_FEAT * FEAT_H * FEAT_W);
    _Float16* cat_h    = (_Float16*)carve(2ULL * BATCH * N_ANCH * D_CAT);       // 62.7 MB
    _Float16* bafT     = (_Float16*)carve(2ULL * BATCH * D_FEAT * N_ANCH);      // 31.4 MB
    _Float16* attn_wh  = (_Float16*)carve(2ULL * N_ANCH * D_FEAT);              //  3.9 MB
    float*    attn_bp  = (float*)   carve(4ULL * N_ANCH);
    _Float16* wcr      = (_Float16*)carve(2ULL * NCR * D_CAT);
    float*    bcr      = (float*)   carve(4ULL * NCR);
    float*    scoreBuf = (float*)   carve(4ULL * N_ANCH * N_ANCH);              // 31.0 MB (reused / batch)
    _Float16* ph       = (_Float16*)carve(2ULL * N_ANCH * N_ANCH);              // 15.5 MB (reused / batch)
    float*    crOut    = (float*)   carve(4ULL * BATCH * N_ANCH * NCR);         //  8.6 MB

    prep_attn_kernel<<<(N_ANCH * D_FEAT + 255) / 256, 256, 0, stream>>>(attn_w, attn_b, attn_wh, attn_bp);
    prep_cr_kernel<<<(NCR * D_CAT + 255) / 256, 256, 0, stream>>>(cls_w, cls_b, reg_w, reg_b, wcr, bcr);
    conv1_kernel<<<BATCH * FEAT_H * FEAT_W, C_FEAT, 0, stream>>>(x, conv1_w, conv1_b, feat);
    gather_kernel<<<BATCH * N_ANCH, 128, 0, stream>>>(feat, cut_xs, invalid, cat_h, bafT);

    const int WPB = 8;   // waves per 256-thread block
    const int MT  = 6;   // 96-row wave tile; 174 M-tiles = 29 * 6
    for (int b = 0; b < BATCH; ++b) {
        // scores = baf(2784x704) * attn_w^T(->2784 cols) + bias, f32 out
        int tilesS = (N_ANCH / (16 * MT)) * (N_ANCH / 32);          // 29*87
        gemm_nt_wmma<false, MT, D_CAT, D_FEAT, N_ANCH>
            <<<(tilesS + WPB - 1) / WPB, 256, 0, stream>>>(
            cat_h + (size_t)b * N_ANCH * D_CAT + D_FEAT,
            attn_wh, scoreBuf,
            N_ANCH / (16 * MT), N_ANCH / 32, D_FEAT, attn_bp);

        softmax_shift_kernel<<<N_ANCH, 256, 0, stream>>>(scoreBuf, ph);

        // attention_features = P(2784x2784) * baf(2784x704)  -> f16 into cat[:,0:704]
        int tilesA = (N_ANCH / (16 * MT)) * (D_FEAT / 32);          // 29*22
        gemm_nt_wmma<true, MT, N_ANCH, N_ANCH, D_CAT>
            <<<(tilesA + WPB - 1) / WPB, 256, 0, stream>>>(
            ph,
            bafT + (size_t)b * D_FEAT * N_ANCH,
            cat_h + (size_t)b * N_ANCH * D_CAT,
            N_ANCH / (16 * MT), D_FEAT / 32, N_ANCH, nullptr);

        // cls+reg = cat(2784x1408) * Wcr^T(96 rows) + bias, f32 out
        int tilesC = (N_ANCH / (16 * MT)) * (NCR / 32);             // 29*3
        gemm_nt_wmma<false, MT, D_CAT, D_CAT, NCR>
            <<<(tilesC + WPB - 1) / WPB, 256, 0, stream>>>(
            cat_h + (size_t)b * N_ANCH * D_CAT,
            wcr,
            crOut + (size_t)b * N_ANCH * NCR,
            N_ANCH / (16 * MT), NCR / 32, D_CAT, bcr);
    }

    assemble_kernel<<<BATCH * N_ANCH, 96, 0, stream>>>(crOut, anchors, out);
}
